// ODEFunc_21560735826058
// MI455X (gfx1250) — matrix operand order repaired and checked by reading, not compiled, not run on hardware
//
#include <hip/hip_runtime.h>
#include <hip/hip_bf16.h>
#include <math.h>

// ---------------------------------------------------------------------------
// CNF dynamics + exact Jacobian trace, restructured into 5 GEMMs:
//   h1 = tanh(x@W1 + b1 + ts*tW)  ; u1 = 1-h1^2          (GEMM1, B x H, K=D)
//   Q  = W2 .* (W1^T @ W3^T)                              (GEMM2, H x H, K=D)
//   h2 = tanh(h1@W2 + b2)         ; u2 = 1-h2^2          (GEMM3, B x H, K=H)
//   dx = h2@W3 + b3                                       (GEMM4, B x D, K=H)
//   V  = u1@Q ; ldj[b] = sum_k V[b,k]*u2[b,k]            (GEMM5 + reduce)
// bf16-split (hi/lo) WMMA with f32 accumulation:
//   A*B ~= Ahi*Bhi + Ahi*Blo + Alo*Bhi   -> ~fp32 accuracy at bf16 WMMA rate.
// K-loop is register-pipelined: next tile's global_load_b128s are issued
// before the WMMA block so HBM/L2 latency hides under matrix math.
// ---------------------------------------------------------------------------

typedef __attribute__((ext_vector_type(16))) __bf16 v16bf;
typedef __attribute__((ext_vector_type(8)))  float  v8f;

#define BM 128
#define BN 128
#define BK 32
#define LDS_K 34   // padded K stride (17 dwords, coprime with 64 banks)

// A fragment (16x32 bf16): lane m=lane&15; lanes<16 -> K 0-7 & 16-23,
// lanes>=16 -> K 8-15 & 24-31 (per CDNA5 ISA 16-bit A-matrix layout).
__device__ __forceinline__ v16bf frag_a(const __bf16* tile, int mrow_base, int lane) {
  const int m  = mrow_base + (lane & 15);
  const int kb = (lane < 16) ? 0 : 8;
  const __bf16* r = tile + m * LDS_K;
  v16bf f;
#pragma unroll
  for (int i = 0; i < 8; ++i) {
    f[i]     = r[kb + i];
    f[i + 8] = r[16 + kb + i];
  }
  return f;
}

// B fragment (32x16 bf16), tile stored TRANSPOSED in LDS as [N][K]:
// lane n=lane&15; lanes<16 -> K 0-7 & 16-23, lanes>=16 -> K 8-15 & 24-31.
__device__ __forceinline__ v16bf frag_b(const __bf16* tile, int ncol_base, int lane) {
  const int n  = ncol_base + (lane & 15);
  const int k0 = (lane < 16) ? 0 : 8;
  const __bf16* r = tile + n * LDS_K + k0;
  v16bf f;
#pragma unroll
  for (int i = 0; i < 8; ++i) { f[i] = r[i]; f[i + 8] = r[16 + i]; }
  return f;
}

// EPI: 0 = C0 = acc (+bias)                      (dx, V)
//      1 = tanh(acc+b+ts*tW) -> C0=h, C1=1-h^2   (h1,u1)
//      2 = tanh(acc+b)       -> C0=h, C1=1-h^2   (h2,u2)
//      3 = C0 = acc * Wq[m,n]                    (Q)
template<int EPI, bool TA, bool TB>
__global__ void __launch_bounds__(256)
gemm_bf16x3(const float* __restrict__ A, const float* __restrict__ B,
            float* __restrict__ C0, float* __restrict__ C1,
            const float* __restrict__ bias, const float* __restrict__ tWv,
            const float* __restrict__ tptr, const float* __restrict__ Wq,
            int M, int N, int K, int lda, int ldb, int ldc)
{
  __shared__ __bf16 sAhi[BM * LDS_K];
  __shared__ __bf16 sAlo[BM * LDS_K];
  __shared__ __bf16 sBhi[BN * LDS_K];
  __shared__ __bf16 sBlo[BN * LDS_K];

  const int tid  = threadIdx.x;
  const int lane = tid & 31;
  const int wid  = tid >> 5;      // 8 waves
  const int wm   = wid >> 2;      // 0..1 : which 64-row half
  const int wn   = wid & 3;       // 0..3 : which 32-col quarter

  const int m0 = blockIdx.y * BM;
  const int n0 = blockIdx.x * BN;

  // --- staging registers: 4 float4 for A tile, 4 float4 for B tile ---
  float4 ra[4], rb[4];

  // Issue global loads for the (m0,n0) tile at K offset kk into ra/rb.
  // Vectorized float4 along whichever index is contiguous in memory.
  auto load_regs = [&](int kk) {
#pragma unroll
    for (int it = 0; it < 4; ++it) {
      const int q = tid + it * 256;                 // float4 index, 1024 total
      if (TA) {        // A(m,k) = A[k*lda + m] ; contiguous in m
        const int k = q >> 5, mv = q & 31;
        ra[it] = *(const float4*)(A + (size_t)(kk + k) * lda + (m0 + 4 * mv));
      } else {         // A(m,k) = A[m*lda + k] ; contiguous in k
        const int m = q >> 3, kv = q & 7;
        ra[it] = *(const float4*)(A + (size_t)(m0 + m) * lda + (kk + 4 * kv));
      }
      if (TB) {        // B(k,n) = B[n*ldb + k] ; contiguous in k
        const int n = q >> 3, kv = q & 7;
        rb[it] = *(const float4*)(B + (size_t)(n0 + n) * ldb + (kk + 4 * kv));
      } else {         // B(k,n) = B[k*ldb + n] ; contiguous in n
        const int k = q >> 5, nv = q & 31;
        rb[it] = *(const float4*)(B + (size_t)(kk + k) * ldb + (n0 + 4 * nv));
      }
    }
  };

  // Split registers into bf16 hi/lo and store to LDS (A:[m][k], B:[n][k]).
  auto store_lds = [&]() {
#pragma unroll
    for (int it = 0; it < 4; ++it) {
      const int q = tid + it * 256;
      const float* av = (const float*)&ra[it];
      const float* bv = (const float*)&rb[it];
#pragma unroll
      for (int j = 0; j < 4; ++j) {
        {
          int m, k;
          if (TA) { k = q >> 5; m = (q & 31) * 4 + j; }
          else    { m = q >> 3; k = (q & 7) * 4 + j;  }
          const __bf16 hi = (__bf16)av[j];
          sAhi[m * LDS_K + k] = hi;
          sAlo[m * LDS_K + k] = (__bf16)(av[j] - (float)hi);
        }
        {
          int n, k;
          if (TB) { n = q >> 3; k = (q & 7) * 4 + j;  }
          else    { k = q >> 5; n = (q & 31) * 4 + j; }
          const __bf16 hi = (__bf16)bv[j];
          sBhi[n * LDS_K + k] = hi;
          sBlo[n * LDS_K + k] = (__bf16)(bv[j] - (float)hi);
        }
      }
    }
  };

  const v8f vzero = {0.f, 0.f, 0.f, 0.f, 0.f, 0.f, 0.f, 0.f};
  v8f acc[4][2];
#pragma unroll
  for (int i = 0; i < 4; ++i)
#pragma unroll
    for (int j = 0; j < 2; ++j) acc[i][j] = vzero;

  // ---- pipeline prologue ----
  load_regs(0);
  store_lds();
  __syncthreads();

  for (int k0 = 0; k0 < K; k0 += BK) {
    const bool has_next = (k0 + BK) < K;
    if (has_next) load_regs(k0 + BK);    // VMEM in flight during WMMAs below
    if (k0 + 2 * BK < K) {               // warm L2 for the tile after next
      __builtin_prefetch(TA ? (A + (size_t)(k0 + 2 * BK) * lda + m0 + tid)
                            : (A + (size_t)(m0 + (tid >> 1)) * lda + k0 + 2 * BK),
                         0, 0);
      __builtin_prefetch(TB ? (B + (size_t)(n0 + (tid >> 1)) * ldb + k0 + 2 * BK)
                            : (B + (size_t)(k0 + 2 * BK) * ldb + n0 + tid),
                         0, 0);
    }

    // ---- compute from LDS ----
    v16bf ah[4], al[4], bh[2], bl[2];
#pragma unroll
    for (int mt = 0; mt < 4; ++mt) {
      ah[mt] = frag_a(sAhi, wm * 64 + mt * 16, lane);
      al[mt] = frag_a(sAlo, wm * 64 + mt * 16, lane);
    }
#pragma unroll
    for (int nt = 0; nt < 2; ++nt) {
      bh[nt] = frag_b(sBhi, wn * 32 + nt * 16, lane);
      bl[nt] = frag_b(sBlo, wn * 32 + nt * 16, lane);
    }
#pragma unroll
    for (int mt = 0; mt < 4; ++mt)
#pragma unroll
      for (int nt = 0; nt < 2; ++nt) {
        acc[mt][nt] = __builtin_amdgcn_wmma_f32_16x16x32_bf16(
            false, ah[mt], false, bh[nt], (short)0, acc[mt][nt], false, false);
        acc[mt][nt] = __builtin_amdgcn_wmma_f32_16x16x32_bf16(
            false, ah[mt], false, bl[nt], (short)0, acc[mt][nt], false, false);
        acc[mt][nt] = __builtin_amdgcn_wmma_f32_16x16x32_bf16(
            false, al[mt], false, bh[nt], (short)0, acc[mt][nt], false, false);
      }

    if (has_next) {
      __syncthreads();   // all waves done reading current LDS tile
      store_lds();       // convert+store next tile (regs already landing)
      __syncthreads();   // stores visible before next compute
    }
  }

  // ---- epilogue (C/D layout: VGPR r -> M = r + (lane<16 ? 0 : 8), N = lane&15) ----
  float ts = 0.0f;
  if (EPI == 1) ts = tptr[0];
  const int col   = lane & 15;
  const int rbase = (lane < 16) ? 0 : 8;
#pragma unroll
  for (int mt = 0; mt < 4; ++mt)
#pragma unroll
    for (int nt = 0; nt < 2; ++nt) {
      const int gn  = n0 + wn * 32 + nt * 16 + col;
      const int gm0 = m0 + wm * 64 + mt * 16 + rbase;
#pragma unroll
      for (int r = 0; r < 8; ++r) {
        const int gm = gm0 + r;
        float c = acc[mt][nt][r];
        if (EPI == 0) {
          if (bias) c += bias[gn];
          C0[(size_t)gm * ldc + gn] = c;
        } else if (EPI == 1) {
          const float h = tanhf(c + bias[gn] + ts * tWv[gn]);
          C0[(size_t)gm * ldc + gn] = h;
          C1[(size_t)gm * ldc + gn] = 1.0f - h * h;
        } else if (EPI == 2) {
          const float h = tanhf(c + bias[gn]);
          C0[(size_t)gm * ldc + gn] = h;
          C1[(size_t)gm * ldc + gn] = 1.0f - h * h;
        } else { // EPI == 3
          C0[(size_t)gm * ldc + gn] = c * Wq[(size_t)gm * ldc + gn];
        }
      }
    }
}

// ldj[b] = sum_k V[b,k]*u2[b,k] ; reg[b] = 0
__global__ void __launch_bounds__(256)
trace_reduce(const float* __restrict__ V, const float* __restrict__ u2,
             float* __restrict__ ldj, float* __restrict__ reg, int Hdim)
{
  const int b = blockIdx.x;
  const float* v = V  + (size_t)b * Hdim;
  const float* u = u2 + (size_t)b * Hdim;
  float s = 0.f;
  for (int k = threadIdx.x; k < Hdim; k += blockDim.x) s += v[k] * u[k];
#pragma unroll
  for (int off = 16; off > 0; off >>= 1) s += __shfl_down(s, off, 32);
  __shared__ float red[8];
  if ((threadIdx.x & 31) == 0) red[threadIdx.x >> 5] = s;
  __syncthreads();
  if (threadIdx.x == 0) {
    float t = 0.f;
#pragma unroll
    for (int i = 0; i < 8; ++i) t += red[i];
    ldj[b] = t;
    reg[b] = 0.f;
  }
}

extern "C" void kernel_launch(void* const* d_in, const int* in_sizes, int n_in,
                              void* d_out, int out_size, void* d_ws, size_t ws_size,
                              hipStream_t stream) {
  const float* x  = (const float*)d_in[0];
  const float* t  = (const float*)d_in[1];
  // d_in[2]=ldj-in (zeros), d_in[3]=reg-in (zeros): unused
  const float* W1 = (const float*)d_in[4];
  const float* b1 = (const float*)d_in[5];
  const float* tW = (const float*)d_in[6];
  const float* W2 = (const float*)d_in[7];
  const float* b2 = (const float*)d_in[8];
  const float* W3 = (const float*)d_in[9];
  const float* b3 = (const float*)d_in[10];

  const int Bn = 1024, Dd = 256, Hh = 1024;
  const size_t BH = (size_t)Bn * Hh, HH = (size_t)Hh * Hh;

  float* ws = (float*)d_ws;
  float* h1 = ws;            // [B,H]
  float* u1 = h1 + BH;       // [B,H]
  float* h2 = u1 + BH;       // [B,H]
  float* u2 = h2 + BH;       // [B,H]
  float* Q  = u2 + BH;       // [H,H]
  float* V  = Q  + HH;       // [B,H]

  float* dx  = (float*)d_out;          // [B,D]
  float* ldj = dx + (size_t)Bn * Dd;   // [B]
  float* reg = ldj + Bn;               // [B]

  const dim3 blk(256);

  // GEMM1: h1/u1 = tanh(x@W1 + b1 + ts*tW)
  gemm_bf16x3<1, false, false><<<dim3(Hh / BN, Bn / BM), blk, 0, stream>>>(
      x, W1, h1, u1, b1, tW, t, nullptr, Bn, Hh, Dd, Dd, Hh, Hh);
  // GEMM2: Q = (W1^T @ W3^T) .* W2     (Pt[j,k] = sum_i W1[i,j]*W3[k,i])
  gemm_bf16x3<3, true, true><<<dim3(Hh / BN, Hh / BM), blk, 0, stream>>>(
      W1, W3, Q, nullptr, nullptr, nullptr, nullptr, W2, Hh, Hh, Dd, Hh, Dd, Hh);
  // GEMM3: h2/u2 = tanh(h1@W2 + b2)
  gemm_bf16x3<2, false, false><<<dim3(Hh / BN, Bn / BM), blk, 0, stream>>>(
      h1, W2, h2, u2, b2, nullptr, nullptr, nullptr, Bn, Hh, Hh, Hh, Hh, Hh);
  // GEMM4: dx = h2@W3 + b3
  gemm_bf16x3<0, false, false><<<dim3(Dd / BN, Bn / BM), blk, 0, stream>>>(
      h2, W3, dx, nullptr, b3, nullptr, nullptr, nullptr, Bn, Dd, Hh, Hh, Dd, Dd);
  // GEMM5: V = u1 @ Q
  gemm_bf16x3<0, false, false><<<dim3(Hh / BN, Bn / BM), blk, 0, stream>>>(
      u1, Q, V, nullptr, nullptr, nullptr, nullptr, nullptr, Bn, Hh, Hh, Hh, Hh, Hh);
  // Reduce: ldj[b] = sum_k V[b,k]*u2[b,k] ; reg = 0
  trace_reduce<<<dim3(Bn), blk, 0, stream>>>(V, u2, ldj, reg, Hh);
}
